// DecoderLayer_70188355551815
// MI455X (gfx1250) — matrix operand order repaired and checked by reading, not compiled, not run on hardware
//
#include <hip/hip_runtime.h>

// CDNA5 / gfx1250 decoder layer. wave32, WMMA f16->f32, TDM async staging.
//  - big GEMMs (M%128==0 && N%128==0): 128x128x32 tiles, 8 waves, A tile staged
//    by TENSOR_LOAD_TO_LDS (double-buffered, TENSORcnt-tracked), B tile per-lane
//    f32->f16 converted into K-contiguous LDS for b128 fragment loads.
//  - small/batched GEMMs (attention heads): proven 64x64x32 WMMA kernel.
//  - MoE: top-2 routed (4x less FLOP than dense reference), gathered-A GEMMs.

typedef __attribute__((ext_vector_type(16))) _Float16 v16h;
typedef __attribute__((ext_vector_type(8)))  float    v8f;
typedef __attribute__((ext_vector_type(4)))  unsigned int u32x4;
typedef __attribute__((ext_vector_type(4)))  int i32x4;
typedef __attribute__((ext_vector_type(8)))  int i32x8;
typedef long long ll;
typedef unsigned long long ull;

#define D_MODEL   1024
#define D_HIDDEN  4096
#define N_LATENTS 64
#define D_LATENT  1024
#define N_HEADS   16
#define N_EXPERTS 8
#define BATCH     2
#define SEQ       2048
#define D_H       64
#define NTOK      (BATCH * SEQ)       // 4096
#define MOE_CHUNK 2048

// ---------------------------------------------------------------------------
// Tensor Data Mover: 2D f32 tile load, global -> LDS, with LDS row padding.
// D# packing per CDNA5 ISA:
//   group0: count=1 | lds_addr | global_addr(57b) | type=2
//   group1: data_size=4B, pad_enable, pad_interval=32 dwords, pad_amount=4 dwords,
//           tensor_dim0=tileK, tensor_dim1=tileRows, tile_dim0=tileK,
//           tile_dim1=tileRows, tensor_dim0_stride=strideElems
// ---------------------------------------------------------------------------
__device__ __forceinline__ void tdm_load_tile_f32(
    const float* gptr, unsigned ldsOff, int tileK, int tileRows, int strideElems) {
  ull ga = (ull)(const void*)gptr;
  u32x4 g0;
  g0[0] = 1u;                                         // count=1 (user mode)
  g0[1] = ldsOff;                                     // lds_addr (bytes)
  g0[2] = (unsigned)ga;                               // global_addr[31:0]
  g0[3] = (unsigned)((ga >> 32) & 0x1ffffffull)       // global_addr[56:32]
        | (2u << 30);                                 // type=2 ("image")
  i32x8 g1;
  g1[0] = (2 << 16)                                   // data_size = 4 bytes
        | (1 << 20)                                   // pad_enable
        | (4 << 22)                                   // pad_interval: 32 dwords
        | (3 << 25);                                  // pad_amount: 4 dwords
  g1[1] = (tileK & 0xffff) << 16;                     // tensor_dim0[15:0]
  g1[2] = (int)(((unsigned)tileK >> 16) | ((unsigned)(tileRows & 0xffff) << 16));
  g1[3] = (int)(((unsigned)tileRows >> 16) | ((unsigned)(tileK & 0xffff) << 16)); // tile_dim0
  g1[4] = tileRows & 0xffff;                          // tile_dim1
  g1[5] = strideElems;                                // tensor_dim0_stride[31:0]
  g1[6] = 0;
  g1[7] = 0;
  i32x4 gz = {0, 0, 0, 0};
#if defined(__clang_major__) && (__clang_major__ >= 23)
  i32x8 gz8 = {0, 0, 0, 0, 0, 0, 0, 0};
  __builtin_amdgcn_tensor_load_to_lds(g0, g1, gz, gz, gz8, 0);
#else
  __builtin_amdgcn_tensor_load_to_lds(g0, g1, gz, gz, 0);
#endif
}

// ---------------------------------------------------------------------------
// Small batched / transposable GEMM: 64x64 macro-tile, 4 waves, 2x2 WMMA each.
// ---------------------------------------------------------------------------
struct GemmArgs {
  const float* A; const float* B; float* C;
  int M, N, K;
  int lda, ldb, ldc;
  int innerCnt;
  ll saO, saI, sbO, sbI, scO, scI;
};

template <bool TRANSB>
__global__ __launch_bounds__(128) void gemm_wmma_kernel(GemmArgs g) {
  constexpr int BM = 64, BN = 64, BK = 32, PAD = 8;
  __shared__ _Float16 As[BM][BK + PAD];   // [m][k]
  __shared__ _Float16 Bs[BN][BK + PAD];   // [n][k] (K-contiguous frag loads)

  const int z     = blockIdx.z;
  const int outer = z / g.innerCnt;
  const int inner = z - outer * g.innerCnt;
  const float* A = g.A + outer * g.saO + inner * g.saI;
  const float* B = g.B + outer * g.sbO + inner * g.sbI;
  float*       C = g.C + outer * g.scO + inner * g.scI;

  const int m0 = blockIdx.y * BM;
  const int n0 = blockIdx.x * BN;

  const int tid  = threadIdx.x;
  const int lane = tid & 31;
  const int wave = tid >> 5;
  const int wm   = (wave >> 1) * 32;
  const int wn   = (wave & 1) * 32;
  const int lrow = lane & 15;
  const int lhi  = lane >> 4;

  const v8f vzero = {0.f, 0.f, 0.f, 0.f, 0.f, 0.f, 0.f, 0.f};
  v8f acc[2][2];
#pragma unroll
  for (int i = 0; i < 2; ++i)
#pragma unroll
    for (int j = 0; j < 2; ++j) acc[i][j] = vzero;

  const int aRow = tid >> 1, aColBase = (tid & 1) * 16;
  const float* aPtr = A + (ll)(m0 + aRow) * g.lda + aColBase;

  for (int k0 = 0; k0 < g.K; k0 += BK) {
    if (k0 + BK < g.K) __builtin_prefetch(aPtr + k0 + BK, 0, 1);  // global_prefetch_b8

#pragma unroll
    for (int i = 0; i < 16; ++i)
      As[aRow][aColBase + i] = (_Float16)aPtr[k0 + i];

    if (TRANSB) {
      const int bN = tid >> 1, bKBase = (tid & 1) * 16;
      const float* bp = B + (ll)(n0 + bN) * g.ldb + k0 + bKBase;
#pragma unroll
      for (int i = 0; i < 16; ++i) Bs[bN][bKBase + i] = (_Float16)bp[i];
    } else {
      const int bK = tid >> 2, bNBase = (tid & 3) * 16;
      const float* bp = B + (ll)(k0 + bK) * g.ldb + n0 + bNBase;
#pragma unroll
      for (int i = 0; i < 16; ++i) Bs[bNBase + i][bK] = (_Float16)bp[i];
    }
    __syncthreads();

    v16h af[2], bf[2];
#pragma unroll
    for (int t = 0; t < 2; ++t) {
      const _Float16* ap0 = &As[wm + t * 16 + lrow][lhi * 16];
      const _Float16* ap1 = &As[wm + t * 16 + lrow][lhi * 16 + 8];
#pragma unroll
      for (int i = 0; i < 8; ++i) { af[t][i] = ap0[i]; af[t][8 + i] = ap1[i]; }
      const _Float16* bp0 = &Bs[wn + t * 16 + lrow][lhi * 16];
#pragma unroll
      for (int i = 0; i < 16; ++i) bf[t][i] = bp0[i];
    }

#pragma unroll
    for (int ti = 0; ti < 2; ++ti)
#pragma unroll
      for (int tj = 0; tj < 2; ++tj)
        acc[ti][tj] = __builtin_amdgcn_wmma_f32_16x16x32_f16(
            false, af[ti], false, bf[tj], (short)0, acc[ti][tj], false, false);
    __syncthreads();
  }

#pragma unroll
  for (int ti = 0; ti < 2; ++ti)
#pragma unroll
    for (int tj = 0; tj < 2; ++tj) {
      float* cp = C + (ll)(m0 + wm + ti * 16 + lhi * 8) * g.ldc +
                  (n0 + wn + tj * 16 + lrow);
#pragma unroll
      for (int v = 0; v < 8; ++v) cp[(ll)v * g.ldc] = acc[ti][tj][v];
    }
}

// ---------------------------------------------------------------------------
// Big GEMM: 128x128 macro-tile, 8 waves, 2x4 WMMA tiles per wave.
// Non-gather: A tile via TDM (double-buffered, issued by wave 0, TENSORcnt).
// Gather (MoE): per-lane A row-gather into the same f32 LDS layout.
// A LDS row stride = 32+4 floats -> matches TDM pad (4 dwords every 32).
// ---------------------------------------------------------------------------
struct GemmBigArgs {
  const float* A; const float* B; float* C;
  int M, N, K;
  int lda, ldb, ldc;
  const int* rowIdx;
  const int* rowCount;
  int rowBase;
};

template <bool GATHER>
__global__ __launch_bounds__(256) void gemm_big_kernel(GemmBigArgs g) {
  constexpr int BM = 128, BN = 128, BK = 32, APAD = 4, BPAD = 8;
  __shared__ float    As[2][BM][BK + APAD];   // f32, TDM target / gather target
  __shared__ _Float16 Bs[BN][BK + BPAD];      // f16, [n][k]

  const int m0 = blockIdx.y * BM;
  const int n0 = blockIdx.x * BN;
  const int mLimit = g.rowCount ? (*g.rowCount - g.rowBase) : g.M;
  if (m0 >= mLimit) return;

  const int tid  = threadIdx.x;
  const int lane = tid & 31;
  const int wave = tid >> 5;
  const int wm   = (wave >> 1) * 32;   // 0,32,64,96
  const int wn   = (wave & 1) * 64;    // 0,64
  const int lrow = lane & 15;
  const int lhi  = lane >> 4;

  const v8f vzero = {0.f, 0.f, 0.f, 0.f, 0.f, 0.f, 0.f, 0.f};
  v8f acc[2][4];
#pragma unroll
  for (int i = 0; i < 2; ++i)
#pragma unroll
    for (int j = 0; j < 4; ++j) acc[i][j] = vzero;

  // gather-mode A loader coords
  const int aRow = tid >> 1, aColBase = (tid & 1) * 16;
  ll aGRow = m0 + aRow;
  if (GATHER) aGRow = g.rowIdx[m0 + aRow];   // list pre-padded with valid zeros
  const float* aPtr = g.A + aGRow * g.lda + aColBase;

  const int nK = g.K / BK;
  if (!GATHER && wave == 0) {
    tdm_load_tile_f32(g.A + (ll)m0 * g.lda,
                      (unsigned)(ull)(const void*)&As[0][0][0],
                      BK, BM, g.lda);
  }

  for (int kt = 0; kt < nK; ++kt) {
    const int buf = kt & 1;
    if (!GATHER) {
      if (wave == 0) {
        if (kt + 1 < nK) {
          tdm_load_tile_f32(g.A + (ll)m0 * g.lda + (ll)(kt + 1) * BK,
                            (unsigned)(ull)(const void*)&As[buf ^ 1][0][0],
                            BK, BM, g.lda);
          __builtin_amdgcn_s_wait_tensorcnt((short)1);  // tile kt complete
        } else {
          __builtin_amdgcn_s_wait_tensorcnt((short)0);
        }
      }
    } else {
      const float* ap = aPtr + kt * BK;
#pragma unroll
      for (int i = 0; i < 16; ++i) As[buf][aRow][aColBase + i] = ap[i];
    }

    {  // B tile: 32x128 f32 -> f16 [n][k]
      const int bK = tid >> 3, bNBase = (tid & 7) * 16;
      const float* bp = g.B + (ll)(kt * BK + bK) * g.ldb + n0 + bNBase;
      if (kt + 1 < nK) __builtin_prefetch(bp + (ll)BK * g.ldb, 0, 1);
#pragma unroll
      for (int i = 0; i < 16; ++i) Bs[bNBase + i][bK] = (_Float16)bp[i];
    }
    __syncthreads();

    v16h af[2], bf[4];
#pragma unroll
    for (int t = 0; t < 2; ++t) {
      const float* ar = &As[buf][wm + t * 16 + lrow][0];
#pragma unroll
      for (int i = 0; i < 8; ++i) {
        af[t][i]     = (_Float16)ar[lhi * 16 + i];
        af[t][8 + i] = (_Float16)ar[lhi * 16 + 8 + i];
      }
    }
#pragma unroll
    for (int t = 0; t < 4; ++t) {
      const _Float16* br = &Bs[wn + t * 16 + lrow][lhi * 16];
#pragma unroll
      for (int i = 0; i < 16; ++i) bf[t][i] = br[i];
    }
#pragma unroll
    for (int ti = 0; ti < 2; ++ti)
#pragma unroll
      for (int tj = 0; tj < 4; ++tj)
        acc[ti][tj] = __builtin_amdgcn_wmma_f32_16x16x32_f16(
            false, af[ti], false, bf[tj], (short)0, acc[ti][tj], false, false);
    __syncthreads();
  }

#pragma unroll
  for (int ti = 0; ti < 2; ++ti)
#pragma unroll
    for (int tj = 0; tj < 4; ++tj) {
      float* cp = g.C + (ll)(m0 + wm + ti * 16 + lhi * 8) * g.ldc +
                  (n0 + wn + tj * 16 + lrow);
#pragma unroll
      for (int v = 0; v < 8; ++v) cp[(ll)v * g.ldc] = acc[ti][tj][v];
    }
}

// ---------------------------------------------------------------------------
// Elementwise / reduction kernels
// ---------------------------------------------------------------------------
__global__ __launch_bounds__(256) void layernorm_kernel(
    const float* __restrict__ X, const float* __restrict__ gw,
    const float* __restrict__ bw, float* __restrict__ Y, int D) {
  const float* x = X + (ll)blockIdx.x * D;
  float* y = Y + (ll)blockIdx.x * D;
  __shared__ float sA[256], sB[256];
  float s = 0.f, q = 0.f;
  for (int i = threadIdx.x; i < D; i += 256) { float v = x[i]; s += v; q += v * v; }
  sA[threadIdx.x] = s; sB[threadIdx.x] = q;
  __syncthreads();
  for (int o = 128; o > 0; o >>= 1) {
    if (threadIdx.x < o) { sA[threadIdx.x] += sA[threadIdx.x + o]; sB[threadIdx.x] += sB[threadIdx.x + o]; }
    __syncthreads();
  }
  const float mean = sA[0] / D;
  const float var  = (sB[0] - sA[0] * sA[0] / D) / (D - 1);   // ddof=1
  const float den  = sqrtf(fmaxf(var, 0.f)) + 1e-6f;
  for (int i = threadIdx.x; i < D; i += 256)
    y[i] = gw[i] * (x[i] - mean) / den + bw[i];
}

__global__ __launch_bounds__(256) void softmax_kernel(float* __restrict__ S, int L, float scale) {
  float* row = S + (ll)blockIdx.x * L;
  __shared__ float sd[256];
  float m = -3.4e38f;
  for (int i = threadIdx.x; i < L; i += 256) m = fmaxf(m, row[i] * scale);
  sd[threadIdx.x] = m; __syncthreads();
  for (int o = 128; o > 0; o >>= 1) {
    if (threadIdx.x < o) sd[threadIdx.x] = fmaxf(sd[threadIdx.x], sd[threadIdx.x + o]);
    __syncthreads();
  }
  m = sd[0]; __syncthreads();
  float sum = 0.f;
  for (int i = threadIdx.x; i < L; i += 256) {
    float e = expf(row[i] * scale - m); row[i] = e; sum += e;
  }
  sd[threadIdx.x] = sum; __syncthreads();
  for (int o = 128; o > 0; o >>= 1) {
    if (threadIdx.x < o) sd[threadIdx.x] += sd[threadIdx.x + o];
    __syncthreads();
  }
  const float inv = 1.f / sd[0];
  for (int i = threadIdx.x; i < L; i += 256) row[i] *= inv;
}

__global__ __launch_bounds__(256) void rope_kernel(
    float* __restrict__ X, const float* __restrict__ cosb,
    const float* __restrict__ sinb, int T, int half, int dh, int dl, ll npairs) {
  ll gid = (ll)blockIdx.x * blockDim.x + threadIdx.x;
  if (gid >= npairs) return;
  const int ppr = dl / 2;
  ll row = gid / ppr;
  int p = (int)(gid - row * ppr);
  int head = p / half, j = p - head * half;
  int t = (int)(row % T);
  int col = head * dh + 2 * j;
  float c = cosb[t * half + j], s = sinb[t * half + j];
  float* xp = X + row * (ll)dl + col;
  float x1 = xp[0], x2 = xp[1];
  xp[0] = x1 * c - x2 * s;
  xp[1] = x1 * s + x2 * c;
}

__global__ __launch_bounds__(256) void router_kernel(
    const float* __restrict__ X, const float* __restrict__ W,
    const float* __restrict__ bias, float* __restrict__ gates,
    int* __restrict__ eidx, int ntok, int D) {
  int gt = blockIdx.x * blockDim.x + threadIdx.x;
  int w = gt >> 5, lane = gt & 31;
  if (w >= ntok) return;
  const float* x = X + (ll)w * D;
  float acc[N_EXPERTS];
#pragma unroll
  for (int e = 0; e < N_EXPERTS; ++e) acc[e] = 0.f;
  for (int i = lane; i < D; i += 32) {
    float xv = x[i];
#pragma unroll
    for (int e = 0; e < N_EXPERTS; ++e) acc[e] += xv * W[i * N_EXPERTS + e];
  }
#pragma unroll
  for (int e = 0; e < N_EXPERTS; ++e)
    for (int o = 16; o > 0; o >>= 1) acc[e] += __shfl_xor(acc[e], o, 32);
  if (lane == 0) {
#pragma unroll
    for (int e = 0; e < N_EXPERTS; ++e) acc[e] += bias[e];
    int i1 = 0;
    for (int e = 1; e < N_EXPERTS; ++e) if (acc[e] > acc[i1]) i1 = e;
    int i2 = (i1 == 0) ? 1 : 0;
    for (int e = 0; e < N_EXPERTS; ++e) if (e != i1 && acc[e] > acc[i2]) i2 = e;
    float e2 = expf(acc[i2] - acc[i1]);
    gates[w * 2]     = 1.f / (1.f + e2);
    gates[w * 2 + 1] = e2 / (1.f + e2);
    eidx[w * 2]      = i1;
    eidx[w * 2 + 1]  = i2;
  }
}

__global__ __launch_bounds__(256) void moe_init_kernel(
    int* __restrict__ counts, int* __restrict__ lists, float* __restrict__ lg, int n) {
  int i = blockIdx.x * blockDim.x + threadIdx.x;
  if (i < N_EXPERTS) counts[i] = 0;
  if (i < n) { lists[i] = 0; lg[i] = 0.f; }
}

__global__ __launch_bounds__(256) void moe_assign_kernel(
    const int* __restrict__ eidx, const float* __restrict__ gates,
    int* __restrict__ counts, int* __restrict__ lists, float* __restrict__ lg, int ntok) {
  int t = blockIdx.x * blockDim.x + threadIdx.x;
  if (t >= ntok) return;
#pragma unroll
  for (int k = 0; k < 2; ++k) {
    int e = eidx[t * 2 + k];
    int pos = atomicAdd(counts + e, 1);
    lists[e * NTOK + pos] = t;
    lg[e * NTOK + pos] = gates[t * 2 + k];
  }
}

__global__ __launch_bounds__(256) void swiglu_kernel(
    float* __restrict__ A, const float* __restrict__ Bv,
    const int* __restrict__ cnt, int rowBase, int H, ll total) {
  ll gid = (ll)blockIdx.x * blockDim.x + threadIdx.x;
  if (gid >= total) return;
  int row = (int)(gid / H);
  if (row + rowBase >= *cnt) return;
  float b = Bv[gid];
  A[gid] = A[gid] * (b / (1.f + expf(-b)));
}

__global__ __launch_bounds__(256) void moe_scatter_kernel(
    const float* __restrict__ Yv, const int* __restrict__ list,
    const float* __restrict__ lg, const int* __restrict__ cnt,
    int rowBase, float* __restrict__ Out, int D, ll total) {
  ll gid = (ll)blockIdx.x * blockDim.x + threadIdx.x;
  if (gid >= total) return;
  int row = (int)(gid / D);
  if (row + rowBase >= *cnt) return;
  int col = (int)(gid - (ll)row * D);
  int t = list[row];
  Out[(ll)t * D + col] += lg[row] * Yv[gid];
}

__global__ __launch_bounds__(256) void zero_kernel(float* __restrict__ p, ll n) {
  ll i = (ll)blockIdx.x * blockDim.x + threadIdx.x;
  if (i < n) p[i] = 0.f;
}

__global__ __launch_bounds__(256) void add2_kernel(
    const float* __restrict__ a, const float* __restrict__ b, float* __restrict__ y, ll n) {
  ll i = (ll)blockIdx.x * blockDim.x + threadIdx.x;
  if (i < n) y[i] = a[i] + b[i];
}

__global__ __launch_bounds__(256) void add3_bias_kernel(
    const float* __restrict__ a, const float* __restrict__ b,
    const float* __restrict__ bias, float* __restrict__ y, int D, ll n) {
  ll i = (ll)blockIdx.x * blockDim.x + threadIdx.x;
  if (i < n) y[i] = a[i] + b[i] + bias[i % D];
}

// ---------------------------------------------------------------------------
// Host-side launchers
// ---------------------------------------------------------------------------
static void gemm(hipStream_t s, const float* A, const float* Bm, float* C,
                 int M, int N, int K, int lda, int ldb, int ldc,
                 bool transb = false, int batch = 1, int inner = 1,
                 ll saO = 0, ll saI = 0, ll sbO = 0, ll sbI = 0, ll scO = 0, ll scI = 0) {
  GemmArgs g{A, Bm, C, M, N, K, lda, ldb, ldc, inner, saO, saI, sbO, sbI, scO, scI};
  dim3 grid(N / 64, M / 64, batch);
  dim3 blk(128);
  if (transb) gemm_wmma_kernel<true><<<grid, blk, 0, s>>>(g);
  else        gemm_wmma_kernel<false><<<grid, blk, 0, s>>>(g);
}

static void gemm_big(hipStream_t s, const float* A, const float* Bm, float* C,
                     int M, int N, int K, int lda, int ldb, int ldc,
                     const int* rowIdx = nullptr, const int* rowCount = nullptr,
                     int rowBase = 0) {
  GemmBigArgs g{A, Bm, C, M, N, K, lda, ldb, ldc, rowIdx, rowCount, rowBase};
  dim3 grid(N / 128, M / 128, 1);
  dim3 blk(256);
  if (rowIdx) gemm_big_kernel<true><<<grid, blk, 0, s>>>(g);
  else        gemm_big_kernel<false><<<grid, blk, 0, s>>>(g);
}

extern "C" void kernel_launch(void* const* d_in, const int* in_sizes, int n_in,
                              void* d_out, int out_size, void* d_ws, size_t ws_size,
                              hipStream_t stream) {
  (void)in_sizes; (void)n_in; (void)out_size; (void)ws_size;
  const float* x      = (const float*)d_in[0];
  const float* cosb   = (const float*)d_in[1];
  const float* sinb   = (const float*)d_in[2];
  const float* ln1g   = (const float*)d_in[3];
  const float* ln1b   = (const float*)d_in[4];
  const float* ln2g   = (const float*)d_in[5];
  const float* ln2b   = (const float*)d_in[6];
  const float* ln3g   = (const float*)d_in[7];
  const float* ln3b   = (const float*)d_in[8];
  const float* Lat    = (const float*)d_in[9];
  const float* Wq_lat = (const float*)d_in[10];
  const float* Wk_in  = (const float*)d_in[11];
  const float* Wv_in  = (const float*)d_in[12];
  const float* Wq_in  = (const float*)d_in[13];
  const float* Wk_lat = (const float*)d_in[14];
  const float* Wv_lat = (const float*)d_in[15];
  const float* Wout   = (const float*)d_in[16];
  const float* rW     = (const float*)d_in[17];
  const float* rb     = (const float*)d_in[18];
  const float* eW     = (const float*)d_in[19];
  const float* eV     = (const float*)d_in[20];
  const float* eWo    = (const float*)d_in[21];
  const float* linW   = (const float*)d_in[22];
  const float* linB   = (const float*)d_in[23];
  float* out = (float*)d_out;

  // ---- workspace layout (floats) -------------------------------------------
  float* wsf = (float*)d_ws;
  ll off = 0;
  auto alloc = [&](ll n) -> float* { float* p = wsf + off; off += (n + 63) & ~63ll; return p; };
  const ll TOK = (ll)NTOK * D_MODEL;                 // 4M
  float* x1   = alloc(TOK);
  float* x2   = alloc(TOK);
  float* moeO = alloc(TOK);
  float* lnt  = alloc(TOK);
  float* Qb   = alloc((ll)N_LATENTS * D_LATENT);
  float* zb   = alloc((ll)BATCH * N_LATENTS * D_LATENT);
  float* Qlb  = alloc((ll)BATCH * N_LATENTS * D_LATENT);
  float* Klb  = alloc((ll)BATCH * N_LATENTS * D_LATENT);
  float* Vlb  = alloc((ll)BATCH * N_LATENTS * D_LATENT);
  float* S2b  = alloc((ll)BATCH * N_HEADS * N_LATENTS * N_LATENTS);
  float* z2b  = alloc((ll)BATCH * N_LATENTS * D_LATENT);
  float* Kzb  = alloc((ll)BATCH * N_LATENTS * D_LATENT);
  float* Vzb  = alloc((ll)BATCH * N_LATENTS * D_LATENT);
  // big union region: attention temps (28M floats) overlaid with MoE scratch
  float* big  = wsf + off;
  float* Kin  = big;
  float* Vin  = big + 1 * TOK;
  float* Qx   = big + 2 * TOK;
  float* S1   = big + 3 * TOK;
  float* S3   = big + 4 * TOK;
  float* xlb  = big + 5 * TOK;
  float* aOut = big + 6 * TOK;
  float* aBuf = big;                                   // MoE: CHUNK x D_HIDDEN
  float* bBuf = big + (ll)MOE_CHUNK * D_HIDDEN;
  float* yBuf = big + 2ll * MOE_CHUNK * D_HIDDEN;      // CHUNK x D_MODEL
  float* linT = big;                                   // post-MoE overlay
  off += 7 * TOK;
  float* gates  = alloc((ll)NTOK * 2);
  float* lgates = alloc((ll)N_EXPERTS * NTOK);
  int* lists  = (int*)alloc((ll)N_EXPERTS * NTOK);
  int* eidx   = (int*)alloc((ll)NTOK * 2);
  int* counts = (int*)alloc(64);

  const int EB = 256;
  const ll HALF = D_H / 2;

  // ================= attention =================
  gemm_big(stream, x, Wk_in, Kin, NTOK, D_LATENT, D_MODEL, 1024, 1024, 1024);
  gemm_big(stream, x, Wv_in, Vin, NTOK, D_LATENT, D_MODEL, 1024, 1024, 1024);
  gemm_big(stream, x, Wq_in, Qx,  NTOK, D_LATENT, D_MODEL, 1024, 1024, 1024);
  {
    ll np = (ll)NTOK * (D_LATENT / 2);
    rope_kernel<<<dim3((np + EB - 1) / EB), dim3(EB), 0, stream>>>(
        Kin, cosb, sinb, SEQ, (int)HALF, D_H, D_LATENT, np);
    rope_kernel<<<dim3((np + EB - 1) / EB), dim3(EB), 0, stream>>>(
        Qx, cosb, sinb, SEQ, (int)HALF, D_H, D_LATENT, np);
  }
  gemm(stream, Lat, Wq_lat, Qb, N_LATENTS, D_LATENT, D_LATENT, 1024, 1024, 1024);

  // S1[b,h] = Q_h @ K_h^T   (Q broadcast over batch)
  gemm(stream, Qb, Kin, S1, N_LATENTS, SEQ, D_H, 1024, 1024, SEQ,
       true, BATCH * N_HEADS, N_HEADS,
       0, 64, (ll)SEQ * 1024, 64, (ll)N_HEADS * N_LATENTS * SEQ, (ll)N_LATENTS * SEQ);
  softmax_kernel<<<dim3(BATCH * N_HEADS * N_LATENTS), dim3(EB), 0, stream>>>(S1, SEQ, 0.125f);
  // z = P1 @ V
  gemm(stream, S1, Vin, zb, N_LATENTS, D_H, SEQ, SEQ, 1024, 1024,
       false, BATCH * N_HEADS, N_HEADS,
       (ll)N_HEADS * N_LATENTS * SEQ, (ll)N_LATENTS * SEQ,
       (ll)SEQ * 1024, 64, (ll)N_LATENTS * 1024, 64);

  gemm(stream, zb, Wq_lat, Qlb, BATCH * N_LATENTS, D_LATENT, D_LATENT, 1024, 1024, 1024);
  gemm(stream, zb, Wk_lat, Klb, BATCH * N_LATENTS, D_LATENT, D_LATENT, 1024, 1024, 1024);
  gemm(stream, zb, Wv_lat, Vlb, BATCH * N_LATENTS, D_LATENT, D_LATENT, 1024, 1024, 1024);

  // S2[b,h] = Ql_h @ Kl_h^T
  gemm(stream, Qlb, Klb, S2b, N_LATENTS, N_LATENTS, D_H, 1024, 1024, N_LATENTS,
       true, BATCH * N_HEADS, N_HEADS,
       (ll)N_LATENTS * 1024, 64, (ll)N_LATENTS * 1024, 64,
       (ll)N_HEADS * N_LATENTS * N_LATENTS, (ll)N_LATENTS * N_LATENTS);
  softmax_kernel<<<dim3(BATCH * N_HEADS * N_LATENTS), dim3(EB), 0, stream>>>(S2b, N_LATENTS, 0.125f);
  gemm(stream, S2b, Vlb, z2b, N_LATENTS, D_H, N_LATENTS, N_LATENTS, 1024, 1024,
       false, BATCH * N_HEADS, N_HEADS,
       (ll)N_HEADS * N_LATENTS * N_LATENTS, (ll)N_LATENTS * N_LATENTS,
       (ll)N_LATENTS * 1024, 64, (ll)N_LATENTS * 1024, 64);

  gemm(stream, z2b, Wk_lat, Kzb, BATCH * N_LATENTS, D_LATENT, D_LATENT, 1024, 1024, 1024);
  gemm(stream, z2b, Wv_lat, Vzb, BATCH * N_LATENTS, D_LATENT, D_LATENT, 1024, 1024, 1024);

  // S3[b,h] = Qx_h @ Kz_h^T
  gemm(stream, Qx, Kzb, S3, SEQ, N_LATENTS, D_H, 1024, 1024, N_LATENTS,
       true, BATCH * N_HEADS, N_HEADS,
       (ll)SEQ * 1024, 64, (ll)N_LATENTS * 1024, 64,
       (ll)N_HEADS * SEQ * N_LATENTS, (ll)SEQ * N_LATENTS);
  softmax_kernel<<<dim3(BATCH * N_HEADS * SEQ), dim3(EB), 0, stream>>>(S3, N_LATENTS, 0.125f);
  gemm(stream, S3, Vzb, xlb, SEQ, D_H, N_LATENTS, N_LATENTS, 1024, 1024,
       false, BATCH * N_HEADS, N_HEADS,
       (ll)N_HEADS * SEQ * N_LATENTS, (ll)SEQ * N_LATENTS,
       (ll)N_LATENTS * 1024, 64, (ll)SEQ * 1024, 64);

  gemm_big(stream, xlb, Wout, aOut, NTOK, D_MODEL, D_LATENT, 1024, 1024, 1024);

  layernorm_kernel<<<dim3(NTOK), dim3(EB), 0, stream>>>(x, ln1g, ln1b, lnt, D_MODEL);
  add2_kernel<<<dim3((TOK + EB - 1) / EB), dim3(EB), 0, stream>>>(lnt, aOut, x1, TOK);

  // ================= MoE (sparse top-2) =================
  router_kernel<<<dim3((NTOK * 32 + EB - 1) / EB), dim3(EB), 0, stream>>>(
      x1, rW, rb, gates, eidx, NTOK, D_MODEL);
  moe_init_kernel<<<dim3((N_EXPERTS * NTOK + EB - 1) / EB), dim3(EB), 0, stream>>>(
      counts, lists, lgates, N_EXPERTS * NTOK);
  zero_kernel<<<dim3((TOK + EB - 1) / EB), dim3(EB), 0, stream>>>(moeO, TOK);
  moe_assign_kernel<<<dim3((NTOK + EB - 1) / EB), dim3(EB), 0, stream>>>(
      eidx, gates, counts, lists, lgates, NTOK);

  for (int e = 0; e < N_EXPERTS; ++e) {
    const float* We = eW + (ll)e * D_MODEL * D_HIDDEN;
    const float* Ve = eV + (ll)e * D_MODEL * D_HIDDEN;
    const float* Wo = eWo + (ll)e * D_HIDDEN * D_MODEL;
    const int* lst = lists + e * NTOK;
    const float* lg = lgates + e * NTOK;
    const int* cnt = counts + e;
    for (int c0 = 0; c0 < NTOK; c0 += MOE_CHUNK) {
      gemm_big(stream, x1, We, aBuf, MOE_CHUNK, D_HIDDEN, D_MODEL, 1024, D_HIDDEN, D_HIDDEN,
               lst + c0, cnt, c0);
      gemm_big(stream, x1, Ve, bBuf, MOE_CHUNK, D_HIDDEN, D_MODEL, 1024, D_HIDDEN, D_HIDDEN,
               lst + c0, cnt, c0);
      ll tot = (ll)MOE_CHUNK * D_HIDDEN;
      swiglu_kernel<<<dim3((tot + EB - 1) / EB), dim3(EB), 0, stream>>>(
          aBuf, bBuf, cnt, c0, D_HIDDEN, tot);
      gemm_big(stream, aBuf, Wo, yBuf, MOE_CHUNK, D_MODEL, D_HIDDEN, D_HIDDEN, 1024, 1024,
               nullptr, cnt, c0);
      ll tot2 = (ll)MOE_CHUNK * D_MODEL;
      moe_scatter_kernel<<<dim3((tot2 + EB - 1) / EB), dim3(EB), 0, stream>>>(
          yBuf, lst + c0, lg + c0, cnt, c0, moeO, D_MODEL, tot2);
    }
  }

  layernorm_kernel<<<dim3(NTOK), dim3(EB), 0, stream>>>(x1, ln2g, ln2b, lnt, D_MODEL);
  add2_kernel<<<dim3((TOK + EB - 1) / EB), dim3(EB), 0, stream>>>(lnt, moeO, x2, TOK);

  // ================= final linear + ln3 =================
  gemm_big(stream, x2, linW, linT, NTOK, D_MODEL, D_MODEL, 1024, 1024, 1024);
  layernorm_kernel<<<dim3(NTOK), dim3(EB), 0, stream>>>(x2, ln3g, ln3b, lnt, D_MODEL);
  add3_bias_kernel<<<dim3((TOK + EB - 1) / EB), dim3(EB), 0, stream>>>(
      lnt, linT, linB, out, D_MODEL, TOK);
}
